// Qwen2MoeSparseMoeBlock_82789789598524
// MI455X (gfx1250) — matrix operand order repaired and checked by reading, not compiled, not run on hardware
//
#include <hip/hip_runtime.h>
#include <cstdint>
#include <cstddef>

typedef __bf16 bf16_t;
typedef __attribute__((ext_vector_type(16))) __bf16 v16bf;
typedef __attribute__((ext_vector_type(8)))  __bf16 v8bf;
typedef __attribute__((ext_vector_type(4)))  __bf16 v4bf;
typedef __attribute__((ext_vector_type(2)))  __bf16 v2bf;
typedef __attribute__((ext_vector_type(8)))  float  v8f;

// ---------------- TDM availability (this toolchain: 6-arg builtin) ----------------
#if defined(__has_builtin)
#  if __has_builtin(__builtin_amdgcn_tensor_load_to_lds) && __has_builtin(__builtin_amdgcn_s_wait_tensorcnt)
#    define HAVE_TDM 1
#  endif
#  if __has_builtin(__builtin_amdgcn_cvt_pk_bf16_f32)
#    define HAVE_PK_BF16 1
#  endif
#  if __has_builtin(__builtin_amdgcn_rcpf)
#    define HAVE_RCPF 1
#  endif
#endif
#ifndef HAVE_TDM
#  define HAVE_TDM 0
#endif
#ifndef HAVE_PK_BF16
#  define HAVE_PK_BF16 0
#endif
#ifndef HAVE_RCPF
#  define HAVE_RCPF 0
#endif

#if HAVE_TDM
typedef __attribute__((ext_vector_type(4))) unsigned int u32x4;
typedef __attribute__((ext_vector_type(8))) int          i32x8;
typedef __attribute__((ext_vector_type(4))) int          i32x4;
#endif

// ---------------- problem constants (from reference) ----------------
constexpr int kM    = 1024;   // tokens
constexpr int kH    = 2048;   // hidden
constexpr int kE    = 60;     // routed experts
constexpr int kI    = 1408;   // moe intermediate
constexpr int kIsh  = 5632;   // shared intermediate
constexpr int kCap  = 1024;   // per-expert token capacity
constexpr int kHRows= 5120;   // padded routed-h rows capacity

// ---------------- GEMM tiling ----------------
constexpr int BM = 64, BN = 64, BK = 32;
constexpr int LDSK = BK + 8;  // 40 elems -> 80B row stride (16B aligned, padded)

__device__ inline uint32_t bfbits(float f) {
  union { float f; uint32_t u; } v; v.f = f;
  return v.u + 0x7FFFu + ((v.u >> 16) & 1u);      // RNE bf16 rounding, keep high 16
}
__device__ inline uint32_t pack2bf(float a, float b) {
#if HAVE_PK_BF16
  v2bf r = __builtin_amdgcn_cvt_pk_bf16_f32(a, b);   // v_cvt_pk_bf16_f32
  return __builtin_bit_cast(uint32_t, r);
#else
  return (bfbits(a) >> 16) | (bfbits(b) & 0xFFFF0000u);
#endif
}
__device__ inline bf16_t f2bf(float f) {
#if HAVE_PK_BF16
  v2bf r = __builtin_amdgcn_cvt_pk_bf16_f32(f, 0.f);
  return r[0];
#else
  uint16_t h = (uint16_t)(bfbits(f) >> 16);
  return __builtin_bit_cast(bf16_t, h);
#endif
}
__device__ inline v8bf cvt8(float4 a, float4 b) {
  union { v8bf v; uint32_t u[4]; } r;
  r.u[0] = pack2bf(a.x, a.y);
  r.u[1] = pack2bf(a.z, a.w);
  r.u[2] = pack2bf(b.x, b.y);
  r.u[3] = pack2bf(b.z, b.w);
  return r.v;
}

// fast sigmoid / silu-mul: v_rcp_f32 instead of IEEE divide
__device__ inline float fast_rcp(float x) {
#if HAVE_RCPF
  return __builtin_amdgcn_rcpf(x);
#else
  return 1.f / x;
#endif
}
__device__ inline float silu_mul(float g, float u) {
  return g * fast_rcp(1.f + __expf(-g)) * u;
}

// low 32 bits of a generic pointer into LDS == LDS byte offset (ISA 10.2 aperture rule)
__device__ inline unsigned lds_off(const void* p) {
  return (unsigned)(unsigned long long)(uintptr_t)p;
}

#if HAVE_TDM
// DMA a BM x BK bf16 tile (rows x 32 elems) global -> LDS with 16B pad per 64B row
// so the LDS layout matches LDSK=40.  remRows rows remain in the tensor from the
// tile start; rows beyond that are zero-filled by TDM OOB handling.
__device__ inline void tdm_load_a(const bf16_t* gsrc, unsigned ldsByte,
                                  int remRows, int ldaElems) {
  unsigned long long ga = (unsigned long long)(uintptr_t)gsrc;
  u32x4 g0;
  g0[0] = 1u;                                   // count=1, user descriptor
  g0[1] = ldsByte;                              // lds_addr (bytes)
  g0[2] = (unsigned)(ga & 0xFFFFFFFFu);         // global_addr[31:0]
  g0[3] = (unsigned)((ga >> 32) & 0x01FFFFFFu) | (2u << 30);  // addr[56:32], type=2
  unsigned td0 = (unsigned)ldaElems;            // tensor dim0 (elements)
  unsigned td1 = (unsigned)remRows;             // tensor dim1 (rows)
  unsigned long long s0 = (unsigned long long)ldaElems;       // dim0 stride (elems)
  i32x8 g1;
  // data_size=1 (2B) @[17:16]; pad_enable @20; pad_interval=3 (64B) @[24:22];
  // pad_amount=3 (4 dwords = 16B) @[31:25]
  g1[0] = (1 << 16) | (1 << 20) | (3 << 22) | (3 << 25);
  g1[1] = (int)((td0 & 0xFFFFu) << 16);                       // dim0[15:0] @[31:16]
  g1[2] = (int)(((td0 >> 16) & 0xFFFFu) | ((td1 & 0xFFFFu) << 16));
  g1[3] = (int)(((td1 >> 16) & 0xFFFFu) | ((unsigned)BK << 16));  // tile_dim0=32
  g1[4] = (int)((unsigned)BM & 0xFFFFu);                      // tile_dim1=64, tile_dim2=0
  g1[5] = (int)(s0 & 0xFFFFFFFFu);
  g1[6] = (int)((s0 >> 32) & 0xFFFFu);                        // dim1_stride unused (2D)
  g1[7] = 0;
  i32x4 z4 = {0, 0, 0, 0};
  i32x8 z8 = {0, 0, 0, 0, 0, 0, 0, 0};
  __builtin_amdgcn_tensor_load_to_lds(g0, g1, z4, z4, z8, 0);
}
#endif

// A fragment: 16x32 bf16. Lanes 0-15: rows M=0..15, K={0..7,16..23};
// lanes 16-31: same rows, K={8..15,24..31}.
__device__ inline v16bf ld_a_frag(const bf16_t* s, int wm, int lane) {
  int m  = wm + (lane & 15);
  int kb = (lane & 16) ? 8 : 0;
  const bf16_t* p = s + m * LDSK + kb;
  union { v16bf v; v8bf h[2]; } r;
  r.h[0] = *(const v8bf*)(p);
  r.h[1] = *(const v8bf*)(p + 16);
  return r.v;
}
// B fragment: 32x16 bf16. Lane holds column N=lane&15; lanes 0-15: K={0..7,16..23},
// lanes 16-31: K={8..15,24..31}.
__device__ inline v16bf ld_b_frag(const bf16_t* s, int cb, int lane) {
  int n  = cb + (lane & 15);
  int kb = (lane & 16) ? 8 : 0;
  const bf16_t* p = s + n * LDSK + kb;
  union { v16bf v; v8bf h[2]; } r;
  r.h[0] = *(const v8bf*)(p);
  r.h[1] = *(const v8bf*)(p + 16);
  return r.v;
}

#define WMMA_BF16(a, b, c) \
  __builtin_amdgcn_wmma_f32_16x16x32_bf16(false, (a), false, (b), (short)0, (c), false, false)

// -------------------------------------------------------------------------
// Modes:
//  0: shared gated-up : Hout = silu(x@Wg.T) * (x@Wu.T), dense rows
//  1: routed gated-up : like 0, A rows gathered via token list of expert z
//  2: shared down     : Fout = (A@B.T) * rowScale[row]
//  3: routed down     : atomicAdd(Fout[token], wt * (A@B.T))
// -------------------------------------------------------------------------
template<int MODE>
__global__ __launch_bounds__(256)
void gemm_wmma_kernel(const bf16_t* __restrict__ A, int ldA, int Mrows,
                      const float* __restrict__ Bf, long long bStride,
                      int K, int upOff,
                      bf16_t* __restrict__ Hout, int ldH,
                      float* __restrict__ Fout, int ldF,
                      const float* __restrict__ rowScale,
                      const int* __restrict__ cnt,
                      const int* __restrict__ offs,
                      const int* __restrict__ toks,
                      const float* __restrict__ wts)
{
  constexpr bool kTDM = (HAVE_TDM != 0) && (MODE != 1);

  __shared__ bf16_t As[2][BM * LDSK];
  __shared__ bf16_t Bs[2][BN * LDSK];
  __shared__ bf16_t Cs[2][BN * LDSK];   // up-projection half (gated modes)

  const int t  = threadIdx.x;
  const int lr = t >> 2;          // tile row handled by this thread for loads
  const int lc = (t & 3) << 3;    // k-segment (0,8,16,24)

  const int e = (MODE == 1 || MODE == 3) ? blockIdx.z : 0;
  int cn = 0, pad = 0, hb = 0;
  if constexpr (MODE == 1 || MODE == 3) {
    cn  = cnt[e];
    pad = (cn + 15) & ~15;
    if ((int)(blockIdx.x * BM) >= pad) return;   // uniform per block
    if constexpr (MODE == 1) hb = offs[e];
  }
  const int m0 = blockIdx.x * BM;
  const int n0 = blockIdx.y * BN;

  const float*  Bbase = Bf + (size_t)e * (size_t)bStride;
  const bf16_t* Ab    = A;
  if constexpr (MODE == 3) Ab = A + (size_t)offs[e] * (size_t)ldA;

  // TDM source: tile start; rows remaining for OOB zero-fill
  const bf16_t* Atile = Ab + (size_t)m0 * ldA;
  int remRows = (MODE == 3) ? (pad - m0) : (Mrows - m0);
  (void)remRows; (void)Atile;

  // Fallback / gather per-thread A row pointer
  bool aval = true;
  const bf16_t* aptr = Ab;
  if constexpr (MODE == 1) {
    int rr = m0 + lr;
    aval = rr < cn;
    int tok = aval ? toks[e * kCap + rr] : 0;
    aptr = Ab + (size_t)tok * ldA;
  } else if constexpr (!kTDM) {
    int rr = m0 + lr;
    if constexpr (MODE == 3) { aval = rr < pad; rr = aval ? rr : 0; }
    aptr = Ab + (size_t)rr * ldA;
  }
  const float* bptr  = Bbase + (size_t)(n0 + lr) * K;
  const float* b2ptr = Bbase + (size_t)(upOff + n0 + lr) * K;

  const int nk = K / BK;
  const v8bf z8 = {};
  v8bf fa = z8, fb = z8, fc = z8;

  // ---------------- prologue: tile 0 ----------------
  if constexpr (kTDM) {
#if HAVE_TDM
    if (t < 32) tdm_load_a(Atile, lds_off(&As[0][0]), remRows, ldA);
#endif
  } else {
    fa = aval ? *(const v8bf*)(aptr + lc) : z8;
    *(v8bf*)&As[0][lr * LDSK + lc] = fa;
  }
  {
    float4 x0 = *(const float4*)(bptr + lc);
    float4 x1 = *(const float4*)(bptr + lc + 4);
    *(v8bf*)&Bs[0][lr * LDSK + lc] = cvt8(x0, x1);
    if constexpr (MODE <= 1) {
      float4 y0 = *(const float4*)(b2ptr + lc);
      float4 y1 = *(const float4*)(b2ptr + lc + 4);
      *(v8bf*)&Cs[0][lr * LDSK + lc] = cvt8(y0, y1);
    }
  }
#if HAVE_TDM
  if constexpr (kTDM) { if (t < 32) __builtin_amdgcn_s_wait_tensorcnt(0); }
#endif
  __syncthreads();

  const int lane = t & 31;
  const int wave = t >> 5;
  const int wm = (wave & 3) * 16;   // 4 waves along M
  const int wn = (wave >> 2) * 32;  // 2 waves along N, 32 cols each

  v8f cg0 = {}, cg1 = {}, cu0 = {}, cu1 = {};

  for (int kt = 0; kt < nk; ++kt) {
    const int  cur  = kt & 1;
    const int  nxt  = cur ^ 1;
    const bool more = (kt + 1) < nk;
    if (more) {
      const int k0 = (kt + 1) * BK;
      if constexpr (kTDM) {
#if HAVE_TDM
        if (t < 32) tdm_load_a(Atile + k0, lds_off(&As[nxt][0]), remRows, ldA);
#endif
      } else {
        fa = aval ? *(const v8bf*)(aptr + k0 + lc) : z8;
      }
      float4 x0 = *(const float4*)(bptr + k0 + lc);
      float4 x1 = *(const float4*)(bptr + k0 + lc + 4);
      fb = cvt8(x0, x1);
      if constexpr (MODE <= 1) {
        float4 y0 = *(const float4*)(b2ptr + k0 + lc);
        float4 y1 = *(const float4*)(b2ptr + k0 + lc + 4);
        fc = cvt8(y0, y1);
      }
      if (kt + 2 < nk) {   // prefetch next-next weight tile into GL2
        __builtin_prefetch(bptr + (size_t)(kt + 2) * BK + lc, 0, 3);
        if constexpr (MODE <= 1)
          __builtin_prefetch(b2ptr + (size_t)(kt + 2) * BK + lc, 0, 3);
      }
    }
    // all fragment loads first, then a dense WMMA burst
    v16bf af  = ld_a_frag(&As[cur][0], wm, lane);
    v16bf bg0 = ld_b_frag(&Bs[cur][0], wn,      lane);
    v16bf bg1 = ld_b_frag(&Bs[cur][0], wn + 16, lane);
    v16bf bu0 = bg0, bu1 = bg1;
    if constexpr (MODE <= 1) {
      bu0 = ld_b_frag(&Cs[cur][0], wn,      lane);
      bu1 = ld_b_frag(&Cs[cur][0], wn + 16, lane);
    }
    cg0 = WMMA_BF16(af, bg0, cg0);
    cg1 = WMMA_BF16(af, bg1, cg1);
    if constexpr (MODE <= 1) {
      cu0 = WMMA_BF16(af, bu0, cu0);
      cu1 = WMMA_BF16(af, bu1, cu1);
    }
    if (more) {
      if constexpr (!kTDM) *(v8bf*)&As[nxt][lr * LDSK + lc] = fa;
      *(v8bf*)&Bs[nxt][lr * LDSK + lc] = fb;
      if constexpr (MODE <= 1) *(v8bf*)&Cs[nxt][lr * LDSK + lc] = fc;
    }
#if HAVE_TDM
    if constexpr (kTDM) { if (more && t < 32) __builtin_amdgcn_s_wait_tensorcnt(0); }
#endif
    __syncthreads();
  }

  // Epilogue. C/D layout: VGPR r -> row wm + r + (lane>=16 ? 8 : 0), col = lane&15.
  const int rbase = wm + ((lane & 16) ? 8 : 0);
  const int cb0 = n0 + wn + (lane & 15);
  const int cb1 = cb0 + 16;
#pragma unroll
  for (int r = 0; r < 8; ++r) {
    const int rr = m0 + rbase + r;
    if constexpr (MODE == 0) {
      Hout[(size_t)rr * ldH + cb0] = f2bf(silu_mul(cg0[r], cu0[r]));
      Hout[(size_t)rr * ldH + cb1] = f2bf(silu_mul(cg1[r], cu1[r]));
    } else if constexpr (MODE == 1) {
      if (rr < pad) {
        Hout[(size_t)(hb + rr) * ldH + cb0] = f2bf(silu_mul(cg0[r], cu0[r]));
        Hout[(size_t)(hb + rr) * ldH + cb1] = f2bf(silu_mul(cg1[r], cu1[r]));
      }
    } else if constexpr (MODE == 2) {
      float s = rowScale[rr];
      Fout[(size_t)rr * ldF + cb0] = cg0[r] * s;
      Fout[(size_t)rr * ldF + cb1] = cg1[r] * s;
    } else { // MODE == 3
      if (rr < cn) {
        int   tok = toks[e * kCap + rr];
        float w   = wts[e * kCap + rr];
        unsafeAtomicAdd(&Fout[(size_t)tok * ldF + cb0], w * cg0[r]);
        unsafeAtomicAdd(&Fout[(size_t)tok * ldF + cb1], w * cg1[r]);
      }
    }
  }
}

// ---------------- router: logits, fp32 softmax, top-4, expert lists ----------------
__global__ __launch_bounds__(64)
void router_kernel(const float* __restrict__ x,
                   const float* __restrict__ gw,
                   const float* __restrict__ sgw,
                   int* __restrict__ cnt, int* __restrict__ toks,
                   float* __restrict__ wts, float* __restrict__ sgate)
{
  __shared__ float xs[kH];
  __shared__ float lg[kE];
  __shared__ float pp[kE];
  const int token = blockIdx.x;
  const float* xr = x + (size_t)token * kH;
  for (int i = threadIdx.x; i < kH; i += 64) xs[i] = xr[i];
  __syncthreads();
  const int tid = threadIdx.x;
  if (tid < kE) {
    const float* w = gw + (size_t)tid * kH;
    float acc = 0.f;
    for (int k = 0; k < kH; k += 4) {
      float4 wv = *(const float4*)(w + k);
      acc += xs[k] * wv.x + xs[k + 1] * wv.y + xs[k + 2] * wv.z + xs[k + 3] * wv.w;
    }
    lg[tid] = acc;
  } else if (tid == kE) {
    float acc = 0.f;
    for (int k = 0; k < kH; k += 4) {
      float4 wv = *(const float4*)(sgw + k);
      acc += xs[k] * wv.x + xs[k + 1] * wv.y + xs[k + 2] * wv.z + xs[k + 3] * wv.w;
    }
    sgate[token] = fast_rcp(1.f + __expf(-acc));
  }
  __syncthreads();
  if (tid == 0) {
    float mx = lg[0];
    for (int i = 1; i < kE; ++i) mx = fmaxf(mx, lg[i]);
    float s = 0.f;
    for (int i = 0; i < kE; ++i) { float p = __expf(lg[i] - mx); pp[i] = p; s += p; }
    float inv = 1.f / s;   // keep precise: feeds routing weights
    for (int k = 0; k < 4; ++k) {           // top-4, first-on-tie like lax.top_k
      int be = 0; float bv = pp[0];
      for (int i = 1; i < kE; ++i) if (pp[i] > bv) { bv = pp[i]; be = i; }
      pp[be] = -1.f;
      int idx = atomicAdd(&cnt[be], 1);
      toks[be * kCap + idx] = token;
      wts[be * kCap + idx]  = bv * inv;     // softmax prob, NO renormalization
    }
  }
}

__global__ __launch_bounds__(256)
void cast_kernel(const float* __restrict__ x, bf16_t* __restrict__ o, int n4) {
  int i = blockIdx.x * 256 + threadIdx.x;
  if (i < n4) {
    float4 v = *(const float4*)(x + 4 * (size_t)i);
    union { v4bf v; uint32_t u[2]; } r;
    r.u[0] = pack2bf(v.x, v.y);
    r.u[1] = pack2bf(v.z, v.w);
    *(v4bf*)(o + 4 * (size_t)i) = r.v;
  }
}

__global__ void zero_cnt_kernel(int* cnt) {
  if (threadIdx.x < kE) cnt[threadIdx.x] = 0;
}

__global__ void scan_kernel(const int* __restrict__ cnt, int* __restrict__ offs) {
  if (threadIdx.x == 0) {
    int o = 0;
    for (int e = 0; e < kE; ++e) { offs[e] = o; o += (cnt[e] + 15) & ~15; }
    offs[kE] = o;
  }
}

// -------------------------------------------------------------------------
extern "C" void kernel_launch(void* const* d_in, const int* in_sizes, int n_in,
                              void* d_out, int out_size, void* d_ws, size_t ws_size,
                              hipStream_t stream) {
  const float* x    = (const float*)d_in[0];  // [M, H]
  const float* gw   = (const float*)d_in[1];  // [E, H]
  const float* sgw  = (const float*)d_in[2];  // [1, H]
  const float* w13  = (const float*)d_in[3];  // [E, 2I, H]
  const float* w2   = (const float*)d_in[4];  // [E, H, I]
  const float* segu = (const float*)d_in[5];  // [2*Ish, H]
  const float* sed  = (const float*)d_in[6];  // [H, Ish]
  float* out = (float*)d_out;                 // [M, H]

  char* w = (char*)d_ws;
  bf16_t* x_bf  = (bf16_t*)w; w += (size_t)kM * kH * sizeof(bf16_t);
  bf16_t* h_sh  = (bf16_t*)w; w += (size_t)kM * kIsh * sizeof(bf16_t);
  bf16_t* h_buf = (bf16_t*)w; w += (size_t)kHRows * kI * sizeof(bf16_t);
  int*    cnt   = (int*)w;    w += 256;
  int*    offs  = (int*)w;    w += 256;
  int*    toks  = (int*)w;    w += (size_t)kE * kCap * sizeof(int);
  float*  wts   = (float*)w;  w += (size_t)kE * kCap * sizeof(float);
  float*  sg    = (float*)w;  w += (size_t)kM * sizeof(float);

  zero_cnt_kernel<<<1, 64, 0, stream>>>(cnt);
  cast_kernel<<<(kM * kH / 4 + 255) / 256, 256, 0, stream>>>(x, x_bf, kM * kH / 4);
  router_kernel<<<kM, 64, 0, stream>>>(x, gw, sgw, cnt, toks, wts, sg);
  scan_kernel<<<1, 32, 0, stream>>>(cnt, offs);

  // shared expert: h_sh = silu(x@Wg.T) * (x@Wu.T)
  gemm_wmma_kernel<0><<<dim3(kM / BM, kIsh / BN, 1), 256, 0, stream>>>(
      x_bf, kH, kM, segu, 0LL, kH, kIsh, h_sh, kIsh, nullptr, 0, nullptr,
      nullptr, nullptr, nullptr, nullptr);

  // shared expert: out = (h_sh @ sed.T) * sigmoid(x@sgw.T)   (writes d_out)
  gemm_wmma_kernel<2><<<dim3(kM / BM, kH / BN, 1), 256, 0, stream>>>(
      h_sh, kIsh, kM, sed, 0LL, kIsh, 0, nullptr, 0, out, kH, sg,
      nullptr, nullptr, nullptr, nullptr);

  // routed experts up: h_buf[offs[e]+r] = silu-mul of gathered tokens
  gemm_wmma_kernel<1><<<dim3(kM / BM, kI / BN, kE), 256, 0, stream>>>(
      x_bf, kH, kM, w13, (long long)(2 * kI) * kH, kH, kI, h_buf, kI, nullptr, 0, nullptr,
      cnt, offs, toks, wts);

  // routed experts down: out[token] += wt * (h @ w2_e.T)   (atomic f32 add)
  gemm_wmma_kernel<3><<<dim3(kM / BM, kH / BN, kE), 256, 0, stream>>>(
      h_buf, kI, 0, w2, (long long)kH * kI, kI, 0, nullptr, 0, out, kH, nullptr,
      cnt, offs, toks, wts);
}
